// LateralInhibition_65223373357690
// MI455X (gfx1250) — matrix operand order repaired and checked by reading, not compiled, not run on hardware
//
#include <hip/hip_runtime.h>

// --- CDNA5 WMMA vector types (probe-confirmed signatures) ---
typedef __attribute__((ext_vector_type(16))) _Float16 v16h;
typedef __attribute__((ext_vector_type(8)))  _Float16 v8h;
typedef __attribute__((ext_vector_type(4)))  _Float16 h4;
typedef __attribute__((ext_vector_type(8)))  float    v8f;
typedef __attribute__((ext_vector_type(4)))  float    f4;

#define NUMC 512          // N == K == 512
#define MB   64           // token rows per workgroup
#define LDA  520          // padded LDS row stride (f16 elements), 1040 B: 16B-aligned, bank-staggered

// Kernel 1: build W^T in f16 with zeroed diagonal.
// wT[n][k] = (n == k) ? 0 : (f16) w[k][n]
__global__ void li_prep_wT(const float* __restrict__ w, _Float16* __restrict__ wT) {
    int idx = blockIdx.x * blockDim.x + threadIdx.x;   // 0 .. 512*512-1
    int n = idx >> 9;
    int k = idx & (NUMC - 1);
    float v = (n == k) ? 0.0f : w[k * NUMC + n];
    wT[idx] = (_Float16)v;
}

// Kernel 2: per-block 64x512 output tile.
//   inhib = X @ W1 + b  (f16 WMMA, f32 accumulate)
//   out   = (inhib > 0) ? X : 0   (exact f32 passthrough of X)
__global__ __launch_bounds__(128, 1) void li_gemm_gate(
    const float*    __restrict__ X,     // (32768, 512) f32
    const _Float16* __restrict__ wT,    // (512, 512) f16, diag zeroed, transposed
    const float*    __restrict__ bias,  // (512,) f32
    float*          __restrict__ out)   // (32768, 512) f32
{
    __shared__ _Float16 sA[MB * LDA];

    const int tid   = threadIdx.x;        // 0..127 (4 waves of 32)
    const int lane  = tid & 31;
    const int wave  = tid >> 5;
    const int lhalf = lane >> 4;          // 0 = lanes 0-15, 1 = lanes 16-31
    const int l15   = lane & 15;
    const long row0 = (long)blockIdx.x * MB;

    // ---- Stage: X block (64x512 f32) -> LDS f16 [64][520] ----
    const f4* Xv = (const f4*)(X + row0 * NUMC);       // 8192 float4 per block
    #pragma unroll 4
    for (int i = 0; i < 64; ++i) {
        int f = tid + i * 128;
        int r = f >> 7;                 // 128 float4 per row
        int c = (f & 127) << 2;
        f4 x = Xv[f];
        h4 hx;
        hx[0] = (_Float16)x[0];
        hx[1] = (_Float16)x[1];
        hx[2] = (_Float16)x[2];
        hx[3] = (_Float16)x[3];
        *(h4*)&sA[r * LDA + c] = hx;    // 8B aligned (c % 4 == 0, 1040 % 8 == 0)
    }
    __syncthreads();

    // ---- Compute: each wave owns 8 N-tiles; process 2 per pass so each
    //      A-fragment fetched from LDS feeds 2 WMMAs ----
    for (int nt = 0; nt < 4; ++nt) {
        const int ncol0 = (wave * 8 + nt * 2) * 16 + l15;   // lane's column, tile 0
        const int ncol1 = ncol0 + 16;                        // lane's column, tile 1
        const _Float16* bcol0 = wT + (size_t)ncol0 * NUMC + lhalf * 16;
        const _Float16* bcol1 = wT + (size_t)ncol1 * NUMC + lhalf * 16;

        v8f acc0[4], acc1[4];
        #pragma unroll
        for (int mt = 0; mt < 4; ++mt) { acc0[mt] = (v8f){}; acc1[mt] = (v8f){}; }

        #pragma unroll 4
        for (int kc = 0; kc < 16; ++kc) {
            const int k = kc * 32;
            // B fragments: 16 contiguous f16 (K = lhalf*16 + 0..15), 32B aligned
            v16h bfrag0 = *(const v16h*)(bcol0 + k);
            v16h bfrag1 = *(const v16h*)(bcol1 + k);
            if (kc + 1 < 16) {
                __builtin_prefetch(bcol0 + k + 32, 0, 3);   // global_prefetch_b8, WGP scope
                __builtin_prefetch(bcol1 + k + 32, 0, 3);
            }

            const int abase = k + lhalf * 16;  // contiguous K per lane half (matches B)
            #pragma unroll
            for (int mt = 0; mt < 4; ++mt) {
                // A fragment: a[0..15] = K abase+0..15
                union { v16h v; struct { v8h lo, hi; } p; } af;
                const int ro = (mt * 16 + l15) * LDA + abase;   // 16B aligned
                af.p.lo = *(const v8h*)&sA[ro];
                af.p.hi = *(const v8h*)&sA[ro + 8];
                acc0[mt] = __builtin_amdgcn_wmma_f32_16x16x32_f16(
                    false, af.v, false, bfrag0, (short)0, acc0[mt], false, false);
                acc1[mt] = __builtin_amdgcn_wmma_f32_16x16x32_f16(
                    false, af.v, false, bfrag1, (short)0, acc1[mt], false, false);
            }
        }

        // ---- Epilogue: bias + hard gate, exact f32 passthrough ----
        const float bb0 = bias[ncol0];
        const float bb1 = bias[ncol1];
        const int mrow0 = lhalf * 8;          // C/D layout: VGPR v -> M = v + 8*lhalf
        #pragma unroll
        for (int mt = 0; mt < 4; ++mt) {
            #pragma unroll
            for (int v = 0; v < 8; ++v) {
                const long r = row0 + mt * 16 + mrow0 + v;
                const size_t off0 = (size_t)r * NUMC + ncol0;
                const size_t off1 = (size_t)r * NUMC + ncol1;
                const float inh0 = acc0[mt][v] + bb0;
                const float inh1 = acc1[mt][v] + bb1;
                out[off0] = (inh0 > 0.0f) ? X[off0] : 0.0f;
                out[off1] = (inh1 > 0.0f) ? X[off1] : 0.0f;
            }
        }
    }
}

extern "C" void kernel_launch(void* const* d_in, const int* in_sizes, int n_in,
                              void* d_out, int out_size, void* d_ws, size_t ws_size,
                              hipStream_t stream) {
    const float* X    = (const float*)d_in[0];   // (8, 4096, 512) f32
    const float* w    = (const float*)d_in[1];   // (512, 512) f32
    const float* bias = (const float*)d_in[2];   // (512,) f32
    float* out = (float*)d_out;

    _Float16* wT = (_Float16*)d_ws;              // 512*512*2 = 512 KB scratch

    // Build transposed, diagonal-zeroed f16 weights.
    li_prep_wT<<<(NUMC * NUMC) / 256, 256, 0, stream>>>(w, wT);

    // 32768 token rows / 64 per block = 512 blocks, 128 threads (4 wave32).
    const int mblocks = (8 * 4096) / MB;
    li_gemm_gate<<<mblocks, 128, 0, stream>>>(X, wT, bias, out);
}
